// EncoderSelfAttention_34961033789518
// MI455X (gfx1250) — compile-verified, matrix operand order checked by reading, not run on hardware
//
#include <hip/hip_runtime.h>
#include <hip/hip_bf16.h>

// ---------------------------------------------------------------------------
// EncoderSelfAttention for MI455X (gfx1250, wave32, WMMA bf16)
//   B=2, T=4096, C=768, H=12, D=64
// Register-blocked (32x64 GEMM tiles / 32-row q-tiles) + explicit fragment
// double-buffering so loads for step i+1 are in flight during step i's WMMAs.
// ---------------------------------------------------------------------------

typedef __attribute__((ext_vector_type(16))) __bf16 v16bf;
typedef __attribute__((ext_vector_type(8)))  float  v8f;

#define B_  2
#define T_  4096
#define C_  768
#define H_  12
#define D_  64
#define C3_ 2304

// D = A(16x32 bf16) x B(32x16 bf16) + C(16x16 f32)
__device__ __forceinline__ v8f wmma_bf16(v16bf a, v16bf b, v8f c) {
    return __builtin_amdgcn_wmma_f32_16x16x32_bf16(
        /*neg_a=*/false, a, /*neg_b=*/false, b,
        /*c_mod=*/(short)0, c, /*reuse_a=*/false, /*reuse_b=*/false);
}

// A/B fragment loader, bf16 row-major source (B-fragment == rows of B^T).
// ISA layout: lane<16 -> row (row0+lane), K = k0+{0..7, 16..23}
//             lane>=16 -> row (row0+lane-16), K = k0+{8..15, 24..31}
__device__ __forceinline__ v16bf load_ab_bf16(const __bf16* base, int row0, int ld, int k0) {
    const int lane = threadIdx.x & 31;
    const int hf   = lane >> 4;
    const long r   = row0 + (lane & 15);
    const __bf16* p = base + r * (long)ld + k0 + hf * 8;
    v16bf f;
#pragma unroll
    for (int i = 0; i < 8; ++i) f[i] = p[i];
#pragma unroll
    for (int i = 0; i < 8; ++i) f[8 + i] = p[16 + i];
    return f;
}

// Same, but fp32 source with on-the-fly bf16 conversion.
__device__ __forceinline__ v16bf load_a_f32(const float* base, int row0, int ld, int k0) {
    const int lane = threadIdx.x & 31;
    const int hf   = lane >> 4;
    const long r   = row0 + (lane & 15);
    const float* p = base + r * (long)ld + k0 + hf * 8;
    v16bf f;
#pragma unroll
    for (int i = 0; i < 8; ++i) f[i]     = (__bf16)p[i];
#pragma unroll
    for (int i = 0; i < 8; ++i) f[8 + i] = (__bf16)p[16 + i];
    return f;
}

// ---------------------------------------------------------------------------
// Kernel 1: W [R][Cc] fp32  ->  Wt [Cc][R] bf16
// ---------------------------------------------------------------------------
__global__ void k_transpose_bf16(const float* __restrict__ W, __bf16* __restrict__ Wt,
                                 int R, int Cc) {
    long idx = (long)blockIdx.x * 256 + threadIdx.x;
    long total = (long)R * Cc;
    if (idx >= total) return;
    int n = (int)(idx / R);
    int c = (int)(idx % R);
    Wt[idx] = (__bf16)W[(long)c * Cc + n];
}

// ---------------------------------------------------------------------------
// Kernel 2: qkv = x @ W_attn + b_attn  ->  Q[b,h,t,d], K[b,h,t,d], Vt[b,h,d,t]
// 32x64 tile per wave; fragments double-buffered across the k-loop.
// ---------------------------------------------------------------------------
__global__ void __launch_bounds__(256, 1) k_qkv(const float*  __restrict__ x,
                                                const __bf16* __restrict__ Wt,   // [3C][C]
                                                const float*  __restrict__ bias,
                                                __bf16* __restrict__ Q,
                                                __bf16* __restrict__ K,
                                                __bf16* __restrict__ Vt) {
    const int warp = threadIdx.x >> 5;
    const int wid  = blockIdx.x * 8 + warp;           // 256*36 waves
    const int tn   = wid % (C3_ / 64);
    const int tm   = wid / (C3_ / 64);
    const int lane = threadIdx.x & 31;
    const int hf = lane >> 4, lc = lane & 15;

    v8f acc[2][4] = {};
    v16bf a0 = load_a_f32(x, tm * 32,      C_, 0);
    v16bf a1 = load_a_f32(x, tm * 32 + 16, C_, 0);
    v16bf b0 = load_ab_bf16(Wt, tn * 64,      C_, 0);
    v16bf b1 = load_ab_bf16(Wt, tn * 64 + 16, C_, 0);
    v16bf b2 = load_ab_bf16(Wt, tn * 64 + 32, C_, 0);
    v16bf b3 = load_ab_bf16(Wt, tn * 64 + 48, C_, 0);

    for (int k0 = 0; k0 < C_; k0 += 32) {
        // Next slab (clamped on last iteration: in-bounds, always defined).
        const int kn = (k0 + 32 < C_) ? k0 + 32 : k0;
        if (k0 + 64 < C_) {
            __builtin_prefetch(x  + (long)(tm * 32 + lc) * C_ + k0 + 64, 0, 3);
            __builtin_prefetch(Wt + (long)(tn * 64 + lc) * C_ + k0 + 64, 0, 3);
        }
        v16bf na0 = load_a_f32(x, tm * 32,      C_, kn);
        v16bf na1 = load_a_f32(x, tm * 32 + 16, C_, kn);
        v16bf nb0 = load_ab_bf16(Wt, tn * 64,      C_, kn);
        v16bf nb1 = load_ab_bf16(Wt, tn * 64 + 16, C_, kn);
        v16bf nb2 = load_ab_bf16(Wt, tn * 64 + 32, C_, kn);
        v16bf nb3 = load_ab_bf16(Wt, tn * 64 + 48, C_, kn);

        acc[0][0] = wmma_bf16(a0, b0, acc[0][0]);
        acc[1][0] = wmma_bf16(a1, b0, acc[1][0]);
        acc[0][1] = wmma_bf16(a0, b1, acc[0][1]);
        acc[1][1] = wmma_bf16(a1, b1, acc[1][1]);
        acc[0][2] = wmma_bf16(a0, b2, acc[0][2]);
        acc[1][2] = wmma_bf16(a1, b2, acc[1][2]);
        acc[0][3] = wmma_bf16(a0, b3, acc[0][3]);
        acc[1][3] = wmma_bf16(a1, b3, acc[1][3]);

        a0 = na0; a1 = na1;
        b0 = nb0; b1 = nb1; b2 = nb2; b3 = nb3;
    }

#pragma unroll
    for (int j = 0; j < 4; ++j) {
        // Sub-tile base is 16-aligned and 768 % 64 == 0 -> which/h/d-base are
        // uniform (scalar) per j; only +lc is per-lane.
        const int nb    = tn * 64 + j * 16;
        const int which = nb / C_;                // 0=Q 1=K 2=V
        const int cnb   = nb % C_;
        const int h     = cnb >> 6;
        const int d     = (cnb & 63) + lc;
        const float bs  = bias[nb + lc];
#pragma unroll
        for (int i = 0; i < 2; ++i) {
#pragma unroll
            for (int r = 0; r < 8; ++r) {
                const int row = tm * 32 + i * 16 + r + hf * 8;  // [0, B*T)
                const int bb = row >> 12;
                const int t  = row & (T_ - 1);
                const __bf16 v = (__bf16)(acc[i][j][r] + bs);
                const long qi = ((long)(bb * H_ + h) * T_ + t) * D_ + d;
                if (which == 0)      Q[qi] = v;
                else if (which == 1) K[qi] = v;
                else                 Vt[((long)(bb * H_ + h) * D_ + d) * T_ + t] = v;
            }
        }
    }
}

// ---------------------------------------------------------------------------
// Kernel 3: flash attention. One wave per (b, h, 32-row q-tile); 32 keys per
// step, online softmax with one q-row per lane (full EXEC). K fragments are
// double-buffered and V loads are issued before the softmax VALU phase, which
// hides their latency. LDS round-trip converts C-layout f32 scores into
// A-layout bf16 P fragments. Same-wave DS ops are in-order on CDNA5 ->
// s_wait_dscnt only, no barriers.
// ---------------------------------------------------------------------------
__global__ void __launch_bounds__(256, 1) k_attn(const __bf16* __restrict__ Q,
                                                 const __bf16* __restrict__ K,
                                                 const __bf16* __restrict__ Vt,
                                                 __bf16* __restrict__ Y) {
    __shared__ __align__(16) float  sS[8][32 * 32];   // raw scores
    __shared__ __align__(16) __bf16 sP[8][32 * 32];   // exp(probs) bf16
    __shared__              float   sA[8][32];        // per-row alpha / 1/l

    const int warp = threadIdx.x >> 5;
    const int lane = threadIdx.x & 31;
    const int hf = lane >> 4, lc = lane & 15;
    const int wid = blockIdx.x * 8 + warp;            // 24*128 waves
    const int qt  = wid & 127;                        // q tile (T/32)
    const int bh  = wid >> 7;                         // 0..23
    const int b   = bh / H_;
    const int h   = bh % H_;

    const __bf16* Qb = Q  + (long)bh * T_ * D_;
    const __bf16* Kb = K  + (long)bh * T_ * D_;
    const __bf16* Vb = Vt + (long)bh * D_ * T_;

    const v16bf qA0 = load_ab_bf16(Qb, qt * 32,      D_, 0);
    const v16bf qA1 = load_ab_bf16(Qb, qt * 32,      D_, 32);
    const v16bf qB0 = load_ab_bf16(Qb, qt * 32 + 16, D_, 0);
    const v16bf qB1 = load_ab_bf16(Qb, qt * 32 + 16, D_, 32);

    v8f oA[4] = {}, oB[4] = {};
    float m = -3.0e38f, l = 0.0f;
    const float scale = 0.125f;                       // 1/sqrt(64)

    float*  S = sS[warp];
    __bf16* P = sP[warp];
    float*  A = sA[warp];

    // Preload K fragments for the first key tile.
    v16bf kf00 = load_ab_bf16(Kb, 0,  D_, 0);
    v16bf kf01 = load_ab_bf16(Kb, 0,  D_, 32);
    v16bf kf10 = load_ab_bf16(Kb, 16, D_, 0);
    v16bf kf11 = load_ab_bf16(Kb, 16, D_, 32);

    for (int k0 = 0; k0 < T_; k0 += 32) {
        // S[32 q, 32 keys] = Q (32x64) @ K^T : 8 WMMAs
        v8f s00 = {}, s01 = {}, s10 = {}, s11 = {};
        s00 = wmma_bf16(qA0, kf00, s00); s00 = wmma_bf16(qA1, kf01, s00);
        s01 = wmma_bf16(qA0, kf10, s01); s01 = wmma_bf16(qA1, kf11, s01);
        s10 = wmma_bf16(qB0, kf00, s10); s10 = wmma_bf16(qB1, kf01, s10);
        s11 = wmma_bf16(qB0, kf10, s11); s11 = wmma_bf16(qB1, kf11, s11);

        // Issue next K tile + current V tile loads; the softmax VALU phase
        // below hides their latency.
        const int kn = (k0 + 32 < T_) ? k0 + 32 : k0;
        v16bf nk00 = load_ab_bf16(Kb, kn,      D_, 0);
        v16bf nk01 = load_ab_bf16(Kb, kn,      D_, 32);
        v16bf nk10 = load_ab_bf16(Kb, kn + 16, D_, 0);
        v16bf nk11 = load_ab_bf16(Kb, kn + 16, D_, 32);
        v16bf vf0 = load_ab_bf16(Vb,  0, T_, k0);
        v16bf vf1 = load_ab_bf16(Vb, 16, T_, k0);
        v16bf vf2 = load_ab_bf16(Vb, 32, T_, k0);
        v16bf vf3 = load_ab_bf16(Vb, 48, T_, k0);
        if (k0 + 64 < T_) {
            __builtin_prefetch(Kb + (long)(k0 + 64 + lc) * D_, 0, 3);
            __builtin_prefetch(Vb + (long)lc * T_ + k0 + 64, 0, 3);
        }

        // C-layout -> LDS row-major [32][32]
#pragma unroll
        for (int r = 0; r < 8; ++r) {
            const int rr = r + hf * 8;
            S[rr * 32 + lc]             = s00[r] * scale;
            S[rr * 32 + 16 + lc]        = s01[r] * scale;
            S[(16 + rr) * 32 + lc]      = s10[r] * scale;
            S[(16 + rr) * 32 + 16 + lc] = s11[r] * scale;
        }
        asm volatile("s_wait_dscnt 0" ::: "memory");

        // Online softmax: lane i owns q-row i (all 32 lanes active).
        {
            const float* Sr = S + lane * 32;
            float mx = m;
#pragma unroll
            for (int j = 0; j < 32; ++j) mx = fmaxf(mx, Sr[j]);
            const float alpha = __expf(m - mx);
            float ls = 0.0f;
#pragma unroll
            for (int j = 0; j < 32; ++j) {
                const float p = __expf(Sr[j] - mx);
                ls += p;
                P[lane * 32 + j] = (__bf16)p;
            }
            l = l * alpha + ls;
            m = mx;
            A[lane] = alpha;
        }
        asm volatile("s_wait_dscnt 0" ::: "memory");

        // Rescale running O by alpha(row).
#pragma unroll
        for (int r = 0; r < 8; ++r) {
            const float aA = A[r + hf * 8];
            const float aB = A[16 + r + hf * 8];
#pragma unroll
            for (int j = 0; j < 4; ++j) { oA[j][r] *= aA; oB[j][r] *= aB; }
        }

        // O[32,64] += P[32,32] @ V[32,64] : 8 WMMAs, V fragments shared.
        v16bf pA = load_ab_bf16(P, 0,  32, 0);
        v16bf pB = load_ab_bf16(P, 16, 32, 0);
        oA[0] = wmma_bf16(pA, vf0, oA[0]);
        oB[0] = wmma_bf16(pB, vf0, oB[0]);
        oA[1] = wmma_bf16(pA, vf1, oA[1]);
        oB[1] = wmma_bf16(pB, vf1, oB[1]);
        oA[2] = wmma_bf16(pA, vf2, oA[2]);
        oB[2] = wmma_bf16(pB, vf2, oB[2]);
        oA[3] = wmma_bf16(pA, vf3, oA[3]);
        oB[3] = wmma_bf16(pB, vf3, oB[3]);

        kf00 = nk00; kf01 = nk01; kf10 = nk10; kf11 = nk11;
    }

    // Normalize by 1/l and store Y[b, t, h*64 + d] bf16.
    A[lane] = 1.0f / l;
    asm volatile("s_wait_dscnt 0" ::: "memory");
#pragma unroll
    for (int i = 0; i < 2; ++i) {
#pragma unroll
        for (int r = 0; r < 8; ++r) {
            const int rr  = i * 16 + r + hf * 8;
            const float inv = A[rr];
            const int t = qt * 32 + rr;
            __bf16* yp = Y + ((long)b * T_ + t) * C_ + h * D_;
#pragma unroll
            for (int j = 0; j < 4; ++j) {
                const float o = (i == 0) ? oA[j][r] : oB[j][r];
                yp[j * 16 + lc] = (__bf16)(o * inv);
            }
        }
    }
}

// ---------------------------------------------------------------------------
// Kernel 4: out = Y @ W_proj + b_proj (fp32 out), 32x64 tile per wave,
// double-buffered fragments.
// ---------------------------------------------------------------------------
__global__ void __launch_bounds__(256, 1) k_proj(const __bf16* __restrict__ Y,
                                                 const __bf16* __restrict__ Wt,  // [C][C]
                                                 const float*  __restrict__ bias,
                                                 float* __restrict__ out) {
    const int warp = threadIdx.x >> 5;
    const int wid  = blockIdx.x * 8 + warp;          // 256*12 waves
    const int tn   = wid % (C_ / 64);
    const int tm   = wid / (C_ / 64);
    const int lane = threadIdx.x & 31;
    const int hf = lane >> 4, lc = lane & 15;

    v8f acc[2][4] = {};
    v16bf a0 = load_ab_bf16(Y, tm * 32,      C_, 0);
    v16bf a1 = load_ab_bf16(Y, tm * 32 + 16, C_, 0);
    v16bf b0 = load_ab_bf16(Wt, tn * 64,      C_, 0);
    v16bf b1 = load_ab_bf16(Wt, tn * 64 + 16, C_, 0);
    v16bf b2 = load_ab_bf16(Wt, tn * 64 + 32, C_, 0);
    v16bf b3 = load_ab_bf16(Wt, tn * 64 + 48, C_, 0);

    for (int k0 = 0; k0 < C_; k0 += 32) {
        const int kn = (k0 + 32 < C_) ? k0 + 32 : k0;
        if (k0 + 64 < C_) {
            __builtin_prefetch(Y  + (long)(tm * 32 + lc) * C_ + k0 + 64, 0, 3);
            __builtin_prefetch(Wt + (long)(tn * 64 + lc) * C_ + k0 + 64, 0, 3);
        }
        v16bf na0 = load_ab_bf16(Y, tm * 32,      C_, kn);
        v16bf na1 = load_ab_bf16(Y, tm * 32 + 16, C_, kn);
        v16bf nb0 = load_ab_bf16(Wt, tn * 64,      C_, kn);
        v16bf nb1 = load_ab_bf16(Wt, tn * 64 + 16, C_, kn);
        v16bf nb2 = load_ab_bf16(Wt, tn * 64 + 32, C_, kn);
        v16bf nb3 = load_ab_bf16(Wt, tn * 64 + 48, C_, kn);

        acc[0][0] = wmma_bf16(a0, b0, acc[0][0]);
        acc[1][0] = wmma_bf16(a1, b0, acc[1][0]);
        acc[0][1] = wmma_bf16(a0, b1, acc[0][1]);
        acc[1][1] = wmma_bf16(a1, b1, acc[1][1]);
        acc[0][2] = wmma_bf16(a0, b2, acc[0][2]);
        acc[1][2] = wmma_bf16(a1, b2, acc[1][2]);
        acc[0][3] = wmma_bf16(a0, b3, acc[0][3]);
        acc[1][3] = wmma_bf16(a1, b3, acc[1][3]);

        a0 = na0; a1 = na1;
        b0 = nb0; b1 = nb1; b2 = nb2; b3 = nb3;
    }
#pragma unroll
    for (int j = 0; j < 4; ++j) {
        const int n = tn * 64 + j * 16 + lc;
        const float bs = bias[n];
#pragma unroll
        for (int i = 0; i < 2; ++i) {
#pragma unroll
            for (int r = 0; r < 8; ++r) {
                const long row = (long)tm * 32 + i * 16 + r + hf * 8;
                out[row * C_ + n] = acc[i][j][r] + bs;
            }
        }
    }
}

// ---------------------------------------------------------------------------
extern "C" void kernel_launch(void* const* d_in, const int* in_sizes, int n_in,
                              void* d_out, int out_size, void* d_ws, size_t ws_size,
                              hipStream_t stream) {
    const float* x      = (const float*)d_in[0];
    const float* W_attn = (const float*)d_in[1];
    const float* b_attn = (const float*)d_in[2];
    const float* W_proj = (const float*)d_in[3];
    const float* b_proj = (const float*)d_in[4];
    float* out = (float*)d_out;
    (void)in_sizes; (void)n_in; (void)out_size; (void)ws_size;

    char* ws = (char*)d_ws;
    size_t off = 0;
    auto alloc = [&](size_t bytes) {
        void* p = ws + off;
        off = (off + bytes + 255) & ~(size_t)255;
        return p;
    };
    __bf16* Wa_t = (__bf16*)alloc((size_t)C3_ * C_ * 2);          // 3.5 MB
    __bf16* Wp_t = (__bf16*)alloc((size_t)C_  * C_ * 2);          // 1.2 MB
    __bf16* Qd   = (__bf16*)alloc((size_t)B_ * H_ * T_ * D_ * 2); // 12.6 MB
    __bf16* Kd   = (__bf16*)alloc((size_t)B_ * H_ * T_ * D_ * 2);
    __bf16* Vtd  = (__bf16*)alloc((size_t)B_ * H_ * T_ * D_ * 2);
    __bf16* Yd   = (__bf16*)alloc((size_t)B_ * T_ * C_ * 2);      // ~55 MB total

    {
        int total = C_ * C3_;
        k_transpose_bf16<<<(total + 255) / 256, 256, 0, stream>>>(W_attn, Wa_t, C_, C3_);
    }
    {
        int total = C_ * C_;
        k_transpose_bf16<<<(total + 255) / 256, 256, 0, stream>>>(W_proj, Wp_t, C_, C_);
    }
    // (B*T/32)*(3C/64) waves / 8 per block
    k_qkv <<< (256 * 36) / 8, 256, 0, stream >>>(x, Wa_t, b_attn, Qd, Kd, Vtd);
    // B*H*(T/32) waves / 8 per block
    k_attn<<< (B_ * H_ * 128) / 8, 256, 0, stream >>>(Qd, Kd, Vtd, Yd);
    // (B*T/32)*(C/64) waves / 8 per block
    k_proj<<< (256 * 12) / 8, 256, 0, stream >>>(Yd, Wp_t, b_proj, out);
}
